// ConvNeXtMambaBlock_76424648065969
// MI455X (gfx1250) — compile-verified, matrix operand order checked by reading, not run on hardware
//
#include <hip/hip_runtime.h>
#include <hip/hip_bf16.h>

// ---------------------------------------------------------------------------
// ConvNeXt + Mamba block for gfx1250 (MI455X), wave32 / WMMA.
// All GEMMs use v_wmma_f32_16x16x32_f16 (f16 in, f32 accumulate).
// ---------------------------------------------------------------------------

typedef __attribute__((ext_vector_type(16))) _Float16 v16h;
typedef __attribute__((ext_vector_type(8)))  float    v8f;

#define DIM      192
#define D_STATE  16
#define D_CONV   4
#define D_INNER  384
#define DT_RANK  12
#define NB       8
#define HH_      32
#define WW_      32
#define LSEQ     1024          // 32*32
#define MROWS    8192          // NB * LSEQ

__device__ __forceinline__ v8f wmma16(v16h a, v16h b, v8f c) {
  return __builtin_amdgcn_wmma_f32_16x16x32_f16(
      /*neg_a=*/false, a, /*neg_b=*/false, b,
      /*c_mod=*/(short)0, c, /*reuse_a=*/false, /*reuse_b=*/false);
}

// ---------------------------------------------------------------------------
// Weight packing: W (K x Nsrc, row-major f32) -> f16 fragments.
// Packed layout: block index (nTile*(K/32)+kb), within block: lane*16+j where
//   lane = (g<<4)|n16 : g selects K half (g*16), n16 = column within tile,
//   j    = K within the 16-run.  value = W[kb*32 + g*16 + j][nTile*16 + n16]
// This makes each lane's 16 halfs contiguous (one v16h load per fragment).
// ---------------------------------------------------------------------------
__global__ void pack_b_kernel(const float* __restrict__ W, _Float16* __restrict__ P,
                              int K, int Nsrc, int Np) {
  int total = Np * K;
  int kb32 = K >> 5;
  for (int idx = blockIdx.x * blockDim.x + threadIdx.x; idx < total;
       idx += gridDim.x * blockDim.x) {
    int blk  = idx >> 9;          // / 512
    int w    = idx & 511;
    int lane = w >> 4;
    int j    = w & 15;
    int kb   = blk % kb32;
    int nT   = blk / kb32;
    int k    = (kb << 5) + ((lane >> 4) << 4) + j;
    int n    = (nT << 4) + (lane & 15);
    P[idx] = (n < Nsrc) ? (_Float16)W[(size_t)k * Nsrc + n] : (_Float16)0.f;
  }
}

// ---------------------------------------------------------------------------
// Fused 7x7 depthwise conv + bias + LayerNorm(eps=1e-6) -> f16 NHWC rows.
// One block per pixel (b,h,w); 192 threads = channels.
// ---------------------------------------------------------------------------
__global__ void dwconv_ln_kernel(const float* __restrict__ x,
                                 const float* __restrict__ w7,   // (7,7,1,192) HWIO
                                 const float* __restrict__ cb,
                                 const float* __restrict__ gw,
                                 const float* __restrict__ gb,
                                 _Float16* __restrict__ normed) {
  __shared__ float ssum[192];
  __shared__ float ssq[192];
  __shared__ float sstat[2];
  int c   = threadIdx.x;
  int bhw = blockIdx.x;
  int b  = bhw >> 10;
  int hw = bhw & 1023;
  int hh = hw >> 5;
  int ww = hw & 31;
  const float* xc = x + (((size_t)b * DIM + c) << 10);   // NCHW channel plane
  float acc = cb[c];
#pragma unroll
  for (int i = 0; i < 7; ++i) {
    int yy = hh - 3 + i;
    if ((unsigned)yy >= 32u) continue;
#pragma unroll
    for (int j = 0; j < 7; ++j) {
      int xx = ww - 3 + j;
      if ((unsigned)xx >= 32u) continue;
      acc += w7[(i * 7 + j) * DIM + c] * xc[(yy << 5) + xx];
    }
  }
  ssum[c] = acc;
  ssq[c]  = acc * acc;
  __syncthreads();
  for (int s = 96; s >= 6; s >>= 1) {
    if (c < s) { ssum[c] += ssum[c + s]; ssq[c] += ssq[c + s]; }
    __syncthreads();
  }
  if (c == 0) {
    float su = 0.f, sq = 0.f;
#pragma unroll
    for (int i = 0; i < 6; ++i) { su += ssum[i]; sq += ssq[i]; }
    float mu  = su * (1.f / 192.f);
    float var = sq * (1.f / 192.f) - mu * mu;
    sstat[0] = mu;
    sstat[1] = rsqrtf(var + 1e-6f);
  }
  __syncthreads();
  float v = (acc - sstat[0]) * sstat[1] * gw[c] + gb[c];
  normed[(size_t)bhw * DIM + c] = (_Float16)v;
}

// ---------------------------------------------------------------------------
// Generic WMMA GEMM: C(MxN) = A(MxK,f16 row-major) * Wpacked + epilogue.
// Block = 128 threads (4 waves); wave tile = 32x32 (2x2 WMMA accumulators).
// grid = (N/32, M/128). M,N,K multiples of 32 guaranteed by launch config,
// so EXEC is all-ones everywhere (WMMA requirement).
// EPI: 1 = f16 store, +bias, exact GELU
//      2 = f16 store, +bias
//      3 = f32 store, no bias
// ---------------------------------------------------------------------------
template <int EPI>
__device__ __forceinline__ void store_tile(void* __restrict__ C,
                                           const float* __restrict__ bias,
                                           v8f acc, int row0, int col0, int N,
                                           int lane) {
  int lm = lane & 15, lg = lane >> 4;
  int col = col0 + lm;
  float bv = (EPI == 1 || EPI == 2) ? bias[col] : 0.f;
#pragma unroll
  for (int r = 0; r < 8; ++r) {
    int row = row0 + (lg << 3) + r;
    float v = acc[r] + bv;
    if (EPI == 1) v = 0.5f * v * (1.f + erff(v * 0.70710678118654752f)); // exact GELU
    if (EPI == 3) ((float*)C)[(size_t)row * N + col] = v;
    else          ((_Float16*)C)[(size_t)row * N + col] = (_Float16)v;
  }
}

template <int EPI>
__global__ void gemm_wmma_kernel(const _Float16* __restrict__ A,
                                 const _Float16* __restrict__ Bp,
                                 const float* __restrict__ bias,
                                 void* __restrict__ C, int M, int N, int K) {
  const int lane = threadIdx.x & 31;
  const int wave = threadIdx.x >> 5;
  const int m0   = ((blockIdx.y << 2) + wave) << 5;
  const int n0   = blockIdx.x << 5;
  const int lm   = lane & 15, lg = lane >> 4;
  const int kb32 = K >> 5;

  const _Float16* a0p = A + (size_t)(m0 + lm) * K + (lg << 4);
  const _Float16* a1p = a0p + (size_t)16 * K;
  const _Float16* b0p = Bp + (size_t)(n0 >> 4) * kb32 * 512 + (lane << 4);
  const _Float16* b1p = b0p + (size_t)kb32 * 512;

  v8f acc00 = {}, acc01 = {}, acc10 = {}, acc11 = {};
  for (int kb = 0; kb < kb32; ++kb) {
    v16h a0 = *(const v16h*)(a0p + (kb << 5));
    v16h a1 = *(const v16h*)(a1p + (kb << 5));
    v16h b0 = *(const v16h*)(b0p + ((size_t)kb << 9));
    v16h b1 = *(const v16h*)(b1p + ((size_t)kb << 9));
    if (kb + 2 < kb32) {                       // uniform branch
      __builtin_prefetch(a0p + ((kb + 2) << 5), 0, 0);   // global_prefetch_b8
      __builtin_prefetch(a1p + ((kb + 2) << 5), 0, 0);
    }
    acc00 = wmma16(a0, b0, acc00);
    acc01 = wmma16(a0, b1, acc01);
    acc10 = wmma16(a1, b0, acc10);
    acc11 = wmma16(a1, b1, acc11);
  }
  store_tile<EPI>(C, bias, acc00, m0,      n0,      N, lane);
  store_tile<EPI>(C, bias, acc01, m0,      n0 + 16, N, lane);
  store_tile<EPI>(C, bias, acc10, m0 + 16, n0,      N, lane);
  store_tile<EPI>(C, bias, acc11, m0 + 16, n0 + 16, N, lane);
}

// ---------------------------------------------------------------------------
// out_proj WMMA GEMM fused with NHWC->NCHW transpose and residual add (f32).
// M rows index (b*1024 + l); N = 192 channels.
// ---------------------------------------------------------------------------
__device__ __forceinline__ void store_res(const float* __restrict__ resid,
                                          float* __restrict__ out, v8f acc,
                                          int row0, int col0, int lane) {
  int lm = lane & 15, lg = lane >> 4;
  int col = col0 + lm;
#pragma unroll
  for (int r = 0; r < 8; ++r) {
    int row = row0 + (lg << 3) + r;
    int b = row >> 10, l = row & 1023;
    size_t oi = (((size_t)b * DIM + col) << 10) + l;
    out[oi] = resid[oi] + acc[r];
  }
}

__global__ void gemm_wmma_residual_kernel(const _Float16* __restrict__ A,
                                          const _Float16* __restrict__ Bp,
                                          const float* __restrict__ resid,
                                          float* __restrict__ out, int K) {
  const int lane = threadIdx.x & 31;
  const int wave = threadIdx.x >> 5;
  const int m0   = ((blockIdx.y << 2) + wave) << 5;
  const int n0   = blockIdx.x << 5;
  const int lm   = lane & 15, lg = lane >> 4;
  const int kb32 = K >> 5;

  const _Float16* a0p = A + (size_t)(m0 + lm) * K + (lg << 4);
  const _Float16* a1p = a0p + (size_t)16 * K;
  const _Float16* b0p = Bp + (size_t)(n0 >> 4) * kb32 * 512 + (lane << 4);
  const _Float16* b1p = b0p + (size_t)kb32 * 512;

  v8f acc00 = {}, acc01 = {}, acc10 = {}, acc11 = {};
  for (int kb = 0; kb < kb32; ++kb) {
    v16h a0 = *(const v16h*)(a0p + (kb << 5));
    v16h a1 = *(const v16h*)(a1p + (kb << 5));
    v16h b0 = *(const v16h*)(b0p + ((size_t)kb << 9));
    v16h b1 = *(const v16h*)(b1p + ((size_t)kb << 9));
    acc00 = wmma16(a0, b0, acc00);
    acc01 = wmma16(a0, b1, acc01);
    acc10 = wmma16(a1, b0, acc10);
    acc11 = wmma16(a1, b1, acc11);
  }
  store_res(resid, out, acc00, m0,      n0,      lane);
  store_res(resid, out, acc01, m0,      n0 + 16, lane);
  store_res(resid, out, acc10, m0 + 16, n0,      lane);
  store_res(resid, out, acc11, m0 + 16, n0 + 16, lane);
}

// ---------------------------------------------------------------------------
// Causal depthwise conv1d (k=4, left pad 3) + SiLU. xm = cols [0,384) of the
// 768-wide in_proj output. Writes f32 (for scan) and f16 (x_proj GEMM A).
// ---------------------------------------------------------------------------
__global__ void conv1d_silu_kernel(const float* __restrict__ xz,
                                   const float* __restrict__ cw,   // (384,4)
                                   const float* __restrict__ cb,
                                   float* __restrict__ xc32,
                                   _Float16* __restrict__ xc16, int total) {
  for (int idx = blockIdx.x * blockDim.x + threadIdx.x; idx < total;
       idx += gridDim.x * blockDim.x) {
    int d    = idx % D_INNER;
    int row  = idx / D_INNER;        // b*1024 + l
    int l    = row & 1023;
    int base = row - l;              // b*1024
    float acc = cb[d];
#pragma unroll
    for (int k = 0; k < D_CONV; ++k) {
      int ll = l - (D_CONV - 1) + k;
      if (ll >= 0) acc += cw[d * D_CONV + k] * xz[(size_t)(base + ll) * 768 + d];
    }
    float s = acc / (1.f + __expf(-acc));   // SiLU
    xc32[idx] = s;
    xc16[idx] = (_Float16)s;
  }
}

// ---------------------------------------------------------------------------
// dt = softplus(dt_raw @ dt_proj_w + dt_proj_b).  K=12 -> scalar FMA.
// proj rows have stride 64 (N padded); dt_raw = cols [0,12).
// ---------------------------------------------------------------------------
__global__ void dt_softplus_kernel(const float* __restrict__ proj,
                                   const float* __restrict__ dtw,  // (12,384)
                                   const float* __restrict__ dtb,
                                   float* __restrict__ dt, int total) {
  for (int idx = blockIdx.x * blockDim.x + threadIdx.x; idx < total;
       idx += gridDim.x * blockDim.x) {
    int d   = idx % D_INNER;
    int row = idx / D_INNER;
    const float* pr = proj + (size_t)row * 64;
    float acc = dtb[d];
#pragma unroll
    for (int r = 0; r < DT_RANK; ++r) acc += pr[r] * dtw[r * D_INNER + d];
    dt[idx] = (acc > 20.f) ? acc : log1pf(__expf(acc));   // softplus
  }
}

// ---------------------------------------------------------------------------
// Selective-scan: h[n] = exp(dt*A[n])*h[n] + dt*B_t[n]*xc ;  y = <h,C_t>.
// Fused: + xc*D, * SiLU(z), f16 output for out_proj GEMM.
// grid (NB, D_INNER/128), 128 threads; each thread owns one channel d with
// h[16] in registers; B_t/C_t staged in LDS per timestep.
// ---------------------------------------------------------------------------
__global__ void ssm_scan_kernel(const float* __restrict__ proj,   // 8192x64
                                const float* __restrict__ dt,     // 8192x384
                                const float* __restrict__ xz,     // 8192x768 (z = cols 384..)
                                const float* __restrict__ xc32,   // 8192x384
                                const float* __restrict__ A_log,  // 384x16
                                const float* __restrict__ Dp,     // 384
                                _Float16* __restrict__ ym) {      // 8192x384
  __shared__ float sBC[2 * D_STATE];
  int b = blockIdx.x;
  int d = blockIdx.y * 128 + threadIdx.x;

  float Ar[D_STATE];
#pragma unroll
  for (int n = 0; n < D_STATE; ++n) Ar[n] = -__expf(A_log[d * D_STATE + n]);
  float h[D_STATE] = {0.f};
  float Dd = Dp[d];

  for (int l = 0; l < LSEQ; ++l) {
    int row = (b << 10) + l;
    if (threadIdx.x < 2 * D_STATE)
      sBC[threadIdx.x] = proj[(size_t)row * 64 + DT_RANK + threadIdx.x];
    __syncthreads();
    float dtv = dt[(size_t)row * D_INNER + d];
    float xcv = xc32[(size_t)row * D_INNER + d];
    float zv  = xz[(size_t)row * 768 + D_INNER + d];
    float dx  = dtv * xcv;
    float acc = 0.f;
#pragma unroll
    for (int n = 0; n < D_STATE; ++n) {
      float hn = __expf(dtv * Ar[n]) * h[n] + sBC[n] * dx;
      h[n] = hn;
      acc += hn * sBC[D_STATE + n];
    }
    float y = acc + xcv * Dd;
    y *= zv / (1.f + __expf(-zv));          // * SiLU(z)
    ym[(size_t)row * D_INNER + d] = (_Float16)y;
    __syncthreads();
  }
}

// ---------------------------------------------------------------------------
// Host-side orchestration.
// ---------------------------------------------------------------------------
extern "C" void kernel_launch(void* const* d_in, const int* in_sizes, int n_in,
                              void* d_out, int out_size, void* d_ws, size_t ws_size,
                              hipStream_t stream) {
  (void)in_sizes; (void)n_in; (void)out_size; (void)ws_size;

  const float* x        = (const float*)d_in[0];   // (8,192,32,32)
  const float* dwconv_w = (const float*)d_in[1];   // (7,7,1,192)
  const float* dwconv_b = (const float*)d_in[2];
  const float* norm_w   = (const float*)d_in[3];
  const float* norm_b   = (const float*)d_in[4];
  const float* pw1_w    = (const float*)d_in[5];   // (192,768)
  const float* pw1_b    = (const float*)d_in[6];
  const float* pw2_w    = (const float*)d_in[7];   // (768,192)
  const float* pw2_b    = (const float*)d_in[8];
  const float* in_proj  = (const float*)d_in[9];   // (192,768)
  const float* conv1d_w = (const float*)d_in[10];  // (384,4)
  const float* conv1d_b = (const float*)d_in[11];
  const float* x_proj   = (const float*)d_in[12];  // (384,44)
  const float* dt_w     = (const float*)d_in[13];  // (12,384)
  const float* dt_b     = (const float*)d_in[14];
  const float* A_log    = (const float*)d_in[15];  // (384,16)
  const float* Dvec     = (const float*)d_in[16];
  const float* out_w    = (const float*)d_in[17];  // (384,192)
  float* out = (float*)d_out;

  // ---- workspace carve-up (256B aligned) ----
  char* base = (char*)d_ws;
  size_t off = 0;
  auto alloc = [&](size_t bytes) -> char* {
    char* p = base + off;
    off = (off + bytes + 255) & ~(size_t)255;
    return p;
  };
  _Float16* p_pw1  = (_Float16*)alloc((size_t)192 * 768 * 2);
  _Float16* p_pw2  = (_Float16*)alloc((size_t)768 * 192 * 2);
  _Float16* p_inp  = (_Float16*)alloc((size_t)192 * 768 * 2);
  _Float16* p_xpr  = (_Float16*)alloc((size_t)384 * 64 * 2);   // N padded 44->64
  _Float16* p_out  = (_Float16*)alloc((size_t)384 * 192 * 2);
  _Float16* normed = (_Float16*)alloc((size_t)MROWS * 192 * 2);
  _Float16* mlp1   = (_Float16*)alloc((size_t)MROWS * 768 * 2);
  _Float16* seq16  = (_Float16*)alloc((size_t)MROWS * 192 * 2);
  float*    xz     = (float*)   alloc((size_t)MROWS * 768 * 4);
  float*    xc32   = (float*)   alloc((size_t)MROWS * 384 * 4);
  _Float16* xc16   = (_Float16*)alloc((size_t)MROWS * 384 * 2);
  float*    projb  = (float*)   alloc((size_t)MROWS * 64 * 4);
  float*    dtbuf  = (float*)   alloc((size_t)MROWS * 384 * 4);
  _Float16* ym16   = (_Float16*)alloc((size_t)MROWS * 384 * 2);

  // ---- 0) pack weights to WMMA fragment layout ----
  auto packgrid = [](int elems) { return dim3((elems + 255) / 256); };
  pack_b_kernel<<<packgrid(768 * 192), 256, 0, stream>>>(pw1_w,   p_pw1, 192, 768, 768);
  pack_b_kernel<<<packgrid(192 * 768), 256, 0, stream>>>(pw2_w,   p_pw2, 768, 192, 192);
  pack_b_kernel<<<packgrid(768 * 192), 256, 0, stream>>>(in_proj, p_inp, 192, 768, 768);
  pack_b_kernel<<<packgrid(64  * 384), 256, 0, stream>>>(x_proj,  p_xpr, 384, 44,  64);
  pack_b_kernel<<<packgrid(192 * 384), 256, 0, stream>>>(out_w,   p_out, 384, 192, 192);

  // ---- 1) dwconv 7x7 + LayerNorm -> normed f16 (8192 x 192) ----
  dwconv_ln_kernel<<<dim3(MROWS), dim3(DIM), 0, stream>>>(
      x, dwconv_w, dwconv_b, norm_w, norm_b, normed);

  // ---- 2) pw1 GEMM + bias + exact GELU -> mlp1 f16 (8192 x 768) ----
  gemm_wmma_kernel<1><<<dim3(768 / 32, MROWS / 128), dim3(128), 0, stream>>>(
      normed, p_pw1, pw1_b, (void*)mlp1, MROWS, 768, 192);

  // ---- 3) pw2 GEMM + bias -> seq f16 (8192 x 192) ----
  gemm_wmma_kernel<2><<<dim3(192 / 32, MROWS / 128), dim3(128), 0, stream>>>(
      mlp1, p_pw2, pw2_b, (void*)seq16, MROWS, 192, 768);

  // ---- 4) in_proj GEMM -> xz f32 (8192 x 768): xm=[0,384), z=[384,768) ----
  gemm_wmma_kernel<3><<<dim3(768 / 32, MROWS / 128), dim3(128), 0, stream>>>(
      seq16, p_inp, nullptr, (void*)xz, MROWS, 768, 192);

  // ---- 5) causal conv1d + SiLU -> xc (f32 + f16) ----
  conv1d_silu_kernel<<<dim3((MROWS * 384 + 255) / 256), dim3(256), 0, stream>>>(
      xz, conv1d_w, conv1d_b, xc32, xc16, MROWS * 384);

  // ---- 6) x_proj GEMM -> proj f32 (8192 x 64; dt|B|C in cols 0..43) ----
  gemm_wmma_kernel<3><<<dim3(64 / 32, MROWS / 128), dim3(128), 0, stream>>>(
      xc16, p_xpr, nullptr, (void*)projb, MROWS, 64, 384);

  // ---- 7) dt = softplus(dt_raw @ dt_proj + b) ----
  dt_softplus_kernel<<<dim3((MROWS * 384 + 255) / 256), dim3(256), 0, stream>>>(
      projb, dt_w, dt_b, dtbuf, MROWS * 384);

  // ---- 8) selective scan + D skip + SiLU(z) gate -> ym f16 ----
  ssm_scan_kernel<<<dim3(NB, D_INNER / 128), dim3(128), 0, stream>>>(
      projb, dtbuf, xz, xc32, A_log, Dvec, ym16);

  // ---- 9) out_proj GEMM + NHWC->NCHW + residual -> d_out f32 ----
  gemm_wmma_residual_kernel<<<dim3(192 / 32, MROWS / 128), dim3(128), 0, stream>>>(
      ym16, p_out, x, out, 384);
}
